// Attention_14113262534930
// MI455X (gfx1250) — compile-verified, matrix operand order checked by reading, not run on hardware
//
#include <hip/hip_runtime.h>
#include <hip/hip_bf16.h>

// MI455X / gfx1250 implementation.
// f16 storage + f32 WMMA accumulation (v_wmma_f32_16x16x32_f16); softmax and
// the cls-bias row-0 fixup in f32. attn (201 MB) is the roofline term; it is
// produced by a fused S->softmax->fixup kernel and consumed exactly once.
// GEMM staging: double-buffered GLOBAL_LOAD_ASYNC_TO_LDS_B128 (ASYNCcnt);
// tile k+1's DMA stays in flight (s_wait_asynccnt 3) while tile k is consumed
// by WMMA — the async engine hides global->LDS latency behind matrix math.

#define NHEADS 12
#define HD 64
#define SEQ 1024
#define BATCH 4
#define CDIM 768

typedef __attribute__((ext_vector_type(16))) _Float16 v16h;
typedef __attribute__((ext_vector_type(8)))  float    v8f;
typedef __attribute__((ext_vector_type(4)))  int      v4i;

#if defined(__has_builtin)
#if __has_builtin(__builtin_amdgcn_global_load_async_to_lds_b128)
#define HAVE_ASYNC_LDS 1
#endif
#endif

__device__ __forceinline__ void async_copy_b128(const _Float16* gsrc, _Float16* lds) {
#if defined(HAVE_ASYNC_LDS)
  __builtin_amdgcn_global_load_async_to_lds_b128(
      (v4i*)const_cast<_Float16*>(gsrc), (v4i*)lds, 0, 0);
#else
  *reinterpret_cast<uint4*>(lds) = *reinterpret_cast<const uint4*>(gsrc);
#endif
}

__device__ __forceinline__ void wait_async_le3() {   // allow next tile in flight
#if defined(HAVE_ASYNC_LDS)
#if __has_builtin(__builtin_amdgcn_s_wait_asynccnt)
  __builtin_amdgcn_s_wait_asynccnt(3);
#else
  asm volatile("s_wait_asynccnt 3" ::: "memory");
#endif
#endif
}

__device__ __forceinline__ void wait_async_zero() {
#if defined(HAVE_ASYNC_LDS)
#if __has_builtin(__builtin_amdgcn_s_wait_asynccnt)
  __builtin_amdgcn_s_wait_asynccnt(0);
#else
  asm volatile("s_wait_asynccnt 0" ::: "memory");
#endif
#endif
}

union FragU { v16h v; unsigned u[8]; _Float16 h[16]; };

// Load a 16x32 f16 WMMA operand tile from a row-major matrix with row stride
// `stride_h` (halves). Pairs fetched as dwords; lane[4] selects K half-groups
// per the CDNA5 16-bit operand layout.
__device__ __forceinline__ v16h load_frag16(const _Float16* base, int row,
                                            int stride_h, int lane) {
  FragU f;
  const int half = (lane >> 4) & 1;
  const unsigned* p = reinterpret_cast<const unsigned*>(base + (size_t)row * stride_h);
#pragma unroll
  for (int q = 0; q < 8; ++q) {
    int kk = ((q < 4) ? (q * 2) : (16 + (q - 4) * 2)) + half * 8;
    f.u[q] = p[kk >> 1];
  }
  return f.v;
}

struct h4 { _Float16 x, y, z, w; };

__global__ __launch_bounds__(256)
void cvt_f32_f16_kernel(const float* __restrict__ src, _Float16* __restrict__ dst, int n4) {
  int i = blockIdx.x * blockDim.x + threadIdx.x;
  if (i < n4) {
    float4 v = reinterpret_cast<const float4*>(src)[i];
    h4 o; o.x = (_Float16)v.x; o.y = (_Float16)v.y; o.z = (_Float16)v.z; o.w = (_Float16)v.w;
    reinterpret_cast<h4*>(dst)[i] = o;
  }
}

// Tiled GEMM: C[M,N] = A[M,K] * W[N,K]^T + bias.
// Block tile 64(M) x 128(N), K-step 32, double-buffered async LDS staging.
// 8 waves as (wm in 0..1) x (wn in 0..3); each wave owns a 32x32 slab ->
// 4 v_wmma_f32_16x16x32_f16 per k-step from 2 A-frags x 2 B-frags.
__global__ __launch_bounds__(256)
void gemm_f16_kernel(const _Float16* __restrict__ A, const _Float16* __restrict__ W,
                     const float* __restrict__ bias, float* __restrict__ Cf,
                     _Float16* __restrict__ Ch, int M, int N, int K) {
  __shared__ _Float16 As[2][64 * 32];
  __shared__ _Float16 Bs[2][128 * 32];
  const int tid  = threadIdx.x;
  const int lane = tid & 31;
  const int wid  = tid >> 5;
  const int wm   = wid & 1;
  const int wn   = wid >> 1;
  const int m0   = blockIdx.y * 64;
  const int n0   = blockIdx.x * 128;

  const int srow = tid >> 2;        // 0..63
  const int sseg = (tid & 3) * 8;   // 16B chunks

  // Per tile, each thread issues exactly 3 b128 async transfers:
  //   A: 64x32 halves  = 256 transfers (1/thread)
  //   B: 128x32 halves = 512 transfers (2/thread, rows r and r+64)
  const _Float16* ga  = A + (size_t)(m0 + srow) * K + sseg;
  const _Float16* gb0 = W + (size_t)(n0 + srow) * K + sseg;
  const _Float16* gb1 = W + (size_t)(n0 + srow + 64) * K + sseg;
  _Float16* la  = &As[0][srow * 32 + sseg];
  _Float16* lb0 = &Bs[0][srow * 32 + sseg];
  _Float16* lb1 = &Bs[0][(srow + 64) * 32 + sseg];
  const int lbuf = 64 * 32;   // As buffer stride (halves)
  const int lbufB = 128 * 32; // Bs buffer stride (halves)

  const int nk = K >> 5;
  // Prologue: tiles 0 and 1 in flight.
  async_copy_b128(ga,       la);
  async_copy_b128(gb0,      lb0);
  async_copy_b128(gb1,      lb1);
  if (nk > 1) {
    async_copy_b128(ga + 32,  la + lbuf);
    async_copy_b128(gb0 + 32, lb0 + lbufB);
    async_copy_b128(gb1 + 32, lb1 + lbufB);
  }

  v8f c00 = {}; v8f c01 = {}; v8f c10 = {}; v8f c11 = {};
  for (int kt = 0; kt < nk; ++kt) {
    if (kt + 1 < nk) wait_async_le3(); else wait_async_zero();
    __syncthreads();
    const int p = kt & 1;
    const _Float16* Ap = &As[p][0];
    const _Float16* Bp = &Bs[p][0];
    v16h a0 = load_frag16(Ap, wm * 32 + (lane & 15), 32, lane);
    v16h a1 = load_frag16(Ap, wm * 32 + 16 + (lane & 15), 32, lane);
    v16h b0 = load_frag16(Bp, wn * 32 + (lane & 15), 32, lane);
    v16h b1 = load_frag16(Bp, wn * 32 + 16 + (lane & 15), 32, lane);
    c00 = __builtin_amdgcn_wmma_f32_16x16x32_f16(false, a0, false, b0, (short)0, c00, false, false);
    c01 = __builtin_amdgcn_wmma_f32_16x16x32_f16(false, a0, false, b1, (short)0, c01, false, false);
    c10 = __builtin_amdgcn_wmma_f32_16x16x32_f16(false, a1, false, b0, (short)0, c10, false, false);
    c11 = __builtin_amdgcn_wmma_f32_16x16x32_f16(false, a1, false, b1, (short)0, c11, false, false);
    __syncthreads();   // everyone done reading buffer p before tile kt+2 overwrites it
    if (kt + 2 < nk) {
      const int koff = (kt + 2) * 32;
      async_copy_b128(ga + koff,  la + p * lbuf);
      async_copy_b128(gb0 + koff, lb0 + p * lbufB);
      async_copy_b128(gb1 + koff, lb1 + p * lbufB);
    }
  }

  const int half = lane >> 4;
  const int nn   = lane & 15;
  const v8f* cc[4] = { &c00, &c01, &c10, &c11 };
#pragma unroll
  for (int t = 0; t < 4; ++t) {
    const int mi = t >> 1, ni = t & 1;
#pragma unroll
    for (int r = 0; r < 8; ++r) {
      int mg = m0 + wm * 32 + mi * 16 + r + 8 * half;
      int ng = n0 + wn * 32 + ni * 16 + nn;
      float v = (*cc[t])[r] + (bias ? bias[ng] : 0.0f);
      if (Cf) Cf[(size_t)mg * N + ng] = v;
      if (Ch) Ch[(size_t)mg * N + ng] = (_Float16)v;
    }
  }
}

// Fused S = scale*Q*K^T -> softmax -> row-0 cls-bias fixup -> attn store.
// One block per (b, h, 16-row slab). 16x1024 f32 score slab in LDS (64 KB).
__global__ __launch_bounds__(256)
void attn_softmax_kernel(const _Float16* __restrict__ qkvh,
                         const float* __restrict__ cls_bias,
                         float* __restrict__ attn) {
  __shared__ float S[16 * SEQ];
  const int rb  = blockIdx.x & 63;
  const int h   = (blockIdx.x >> 6) % NHEADS;
  const int b   = blockIdx.x / (64 * NHEADS);
  const int tid = threadIdx.x;
  const int lane = tid & 31;
  const int wid  = tid >> 5;

  const size_t qkvBase = (size_t)b * SEQ * (3 * CDIM);
  const _Float16* qbase = qkvh + qkvBase + h * HD;
  const _Float16* kbase = qkvh + qkvBase + (NHEADS + h) * HD;

  const int mrow = rb * 16 + (lane & 15);
  v16h a0 = load_frag16(qbase,      mrow, 3 * CDIM, lane);
  v16h a1 = load_frag16(qbase + 32, mrow, 3 * CDIM, lane);

  const float scale = 0.125f;  // hd^-0.5
  const int half = lane >> 4;
#pragma unroll
  for (int nt = 0; nt < 8; ++nt) {
    const int ncol0 = wid * 128 + nt * 16;
    v16h b0 = load_frag16(kbase,      ncol0 + (lane & 15), 3 * CDIM, lane);
    v16h b1 = load_frag16(kbase + 32, ncol0 + (lane & 15), 3 * CDIM, lane);
    v8f c = {};
    c = __builtin_amdgcn_wmma_f32_16x16x32_f16(false, a0, false, b0, (short)0, c, false, false);
    c = __builtin_amdgcn_wmma_f32_16x16x32_f16(false, a1, false, b1, (short)0, c, false, false);
    const int nn = ncol0 + (lane & 15);
#pragma unroll
    for (int r = 0; r < 8; ++r) S[(r + 8 * half) * SEQ + nn] = c[r] * scale;
  }
  __syncthreads();

  // Row softmax: 16 threads per row; shfl_xor(width=16) reductions stay in
  // contiguous 16-lane groups of a wave32.
  const int row = tid >> 4;
  const int j   = tid & 15;
  float* Sr = S + row * SEQ + j * 64;
  float mx = -3.4e38f;
  for (int c2 = 0; c2 < 64; ++c2) mx = fmaxf(mx, Sr[c2]);
  for (int off = 8; off; off >>= 1) mx = fmaxf(mx, __shfl_xor(mx, off, 16));
  float sm = 0.0f;
  for (int c2 = 0; c2 < 64; ++c2) { float e = __expf(Sr[c2] - mx); Sr[c2] = e; sm += e; }
  for (int off = 8; off; off >>= 1) sm += __shfl_xor(sm, off, 16);
  const float inv = 1.0f / sm;
  for (int c2 = 0; c2 < 64; ++c2) Sr[c2] *= inv;
  __syncthreads();

  if (rb == 0 && tid < 16) {
    float p00 = S[0];
    float part = 0.0f;
    for (int c2 = 0; c2 < 64; ++c2) { int col = j * 64 + c2; if (col) part += S[col]; }
    for (int off = 8; off; off >>= 1) part += __shfl_xor(part, off, 16);
    float a00 = fminf(fmaxf(p00 + cls_bias[h], 0.0f), 1.0f);
    float mp  = (1.0f - a00) / (part + 1e-6f);
    for (int c2 = 0; c2 < 64; ++c2) {
      int col = j * 64 + c2;
      S[col] = (col == 0) ? a00 : fminf(fmaxf(S[col] * mp, 0.0f), 1.0f);
    }
  }
  __syncthreads();

  float4* dst = reinterpret_cast<float4*>(
      attn + (size_t)((b * NHEADS + h) * SEQ + rb * 16 + row) * SEQ + j * 64);
  const float4* srcv = reinterpret_cast<const float4*>(Sr);
#pragma unroll
  for (int c2 = 0; c2 < 16; ++c2) dst[c2] = srcv[c2];
}

// out_h[b,h] = attn[b,h] (1024x1024) @ V[b,h] (1024x64).
__global__ __launch_bounds__(256)
void attn_v_kernel(const float* __restrict__ attn, const _Float16* __restrict__ qkvh,
                   _Float16* __restrict__ aoh) {
  __shared__ _Float16 As[64 * 32];
  __shared__ _Float16 Vt[64 * 32];  // [d][k] transposed V tile
  const int rb  = blockIdx.x & 15;
  const int h   = (blockIdx.x >> 4) % NHEADS;
  const int b   = blockIdx.x / (16 * NHEADS);
  const int tid = threadIdx.x;
  const int lane = tid & 31;
  const int wid  = tid >> 5;
  const int mt   = wid & 3;
  const int nt0  = (wid >> 2) * 2;
  const int m0   = rb * 64;

  const float* arow = attn + (size_t)((b * NHEADS + h) * SEQ + m0) * SEQ;
  const _Float16* vbase = qkvh + (size_t)b * SEQ * (3 * CDIM) + (2 * NHEADS + h) * HD;

  v8f c0 = {}; v8f c1 = {};
  for (int k0 = 0; k0 < SEQ; k0 += 32) {
#pragma unroll
    for (int i = tid; i < 2048; i += 256) {          // attn 64x32 -> f16
      int r = i >> 5, cc = i & 31;
      As[i] = (_Float16)arow[(size_t)r * SEQ + k0 + cc];
    }
#pragma unroll
    for (int i = tid; i < 2048; i += 256) {          // V^T 64x32
      int kk = i >> 6, d = i & 63;
      Vt[d * 32 + kk] = vbase[(size_t)(k0 + kk) * (3 * CDIM) + d];
    }
    __syncthreads();
    v16h a  = load_frag16(As, mt * 16 + (lane & 15), 32, lane);
    v16h b0 = load_frag16(Vt, nt0 * 16 + (lane & 15), 32, lane);
    v16h b1 = load_frag16(Vt, (nt0 + 1) * 16 + (lane & 15), 32, lane);
    c0 = __builtin_amdgcn_wmma_f32_16x16x32_f16(false, a, false, b0, (short)0, c0, false, false);
    c1 = __builtin_amdgcn_wmma_f32_16x16x32_f16(false, a, false, b1, (short)0, c1, false, false);
    __syncthreads();
  }
  const int half = lane >> 4;
  const int nn   = lane & 15;
#pragma unroll
  for (int r = 0; r < 8; ++r) {
    int m = m0 + mt * 16 + r + 8 * half;
    size_t rowoff = ((size_t)b * SEQ + m) * CDIM + h * HD;
    aoh[rowoff + nt0 * 16 + nn]       = (_Float16)c0[r];
    aoh[rowoff + (nt0 + 1) * 16 + nn] = (_Float16)c1[r];
  }
}

extern "C" void kernel_launch(void* const* d_in, const int* in_sizes, int n_in,
                              void* d_out, int out_size, void* d_ws, size_t ws_size,
                              hipStream_t stream) {
  const float* x      = (const float*)d_in[0];
  const float* qkv_w  = (const float*)d_in[1];
  const float* qkv_b  = (const float*)d_in[2];
  const float* proj_w = (const float*)d_in[3];
  const float* proj_b = (const float*)d_in[4];
  const float* cls_b  = (const float*)d_in[5];

  float* out  = (float*)d_out;                       // (B,N,C) f32
  float* attn = out + (size_t)BATCH * SEQ * CDIM;    // (B,H,N,N) f32

  _Float16* xh   = (_Float16*)d_ws;                          // 4096 x 768
  _Float16* wh   = xh   + (size_t)BATCH * SEQ * CDIM;        // 2304 x 768
  _Float16* pwh  = wh   + (size_t)3 * CDIM * CDIM;           // 768 x 768
  _Float16* qkvh = pwh  + (size_t)CDIM * CDIM;               // 4096 x 2304
  _Float16* aoh  = qkvh + (size_t)BATCH * SEQ * 3 * CDIM;    // 4096 x 768

  const int nx = BATCH * SEQ * CDIM / 4;
  const int nw = 3 * CDIM * CDIM / 4;
  const int np = CDIM * CDIM / 4;
  cvt_f32_f16_kernel<<<(nx + 255) / 256, 256, 0, stream>>>(x, xh, nx);
  cvt_f32_f16_kernel<<<(nw + 255) / 256, 256, 0, stream>>>(qkv_w, wh, nw);
  cvt_f32_f16_kernel<<<(np + 255) / 256, 256, 0, stream>>>(proj_w, pwh, np);

  gemm_f16_kernel<<<dim3(3 * CDIM / 128, BATCH * SEQ / 64), 256, 0, stream>>>(
      xh, wh, qkv_b, nullptr, qkvh, BATCH * SEQ, 3 * CDIM, CDIM);

  attn_softmax_kernel<<<BATCH * NHEADS * (SEQ / 16), 256, 0, stream>>>(qkvh, cls_b, attn);

  attn_v_kernel<<<BATCH * NHEADS * (SEQ / 64), 256, 0, stream>>>(attn, qkvh, aoh);

  gemm_f16_kernel<<<dim3(CDIM / 128, BATCH * SEQ / 64), 256, 0, stream>>>(
      aoh, pwh, proj_b, out, nullptr, BATCH * SEQ, CDIM, CDIM);
}
